// ComputeFBANK_18915035972198
// MI455X (gfx1250) — matrix operand order repaired and checked by reading, not compiled, not run on hardware
//
#include <hip/hip_runtime.h>
#include <hip/hip_bf16.h>
#include <math.h>

typedef _Float16 v16h __attribute__((ext_vector_type(16)));
typedef _Float16 h8   __attribute__((ext_vector_type(8)));
typedef float    v8f  __attribute__((ext_vector_type(8)));

#define BATCH   64
#define LEN     240000
#define WIN     400
#define SHIFT   160
#define NFR     1498          // 1 + (240000-400)/160
#define NMEL    80
#define FT_TILES 94           // ceil(1498/16)
#define KP      416           // padded frame length (13*32)
#define KSTEPS  13
#define NBT     13            // bin tiles -> 208 bins (cols >=200 zeroed)
#define SPECW   224           // padded spectrum width for mel GEMM K (7*32)
#define SKSTEPS 7
#define MELT    5             // 80 mels = 5 tiles of 16
#define FEPS    2.2204460492503131e-16f

// workspace layout (bytes)
#define WS_T_OFF   0                      // 64 ints
#define WS_COS_OFF 256                    // 13*13*512 halves = 173056 B
#define WS_SIN_OFF (256 + 173056)
#define WS_BNK_OFF (256 + 2*173056)       // 7*5*512 halves = 35840 B

// ---------------------------------------------------------------------------
// T_ = trunc(T / (max(T)/80)) exactly as the reference (f32 ops, trunc to i32)
__global__ void k_prepT(const int* __restrict__ T, int* __restrict__ Tc) {
  __shared__ int mx;
  if (threadIdx.x == 0) {
    int m = T[0];
    for (int i = 1; i < BATCH; ++i) m = max(m, T[i]);
    mx = m;
  }
  __syncthreads();
  float ds = (float)mx / 80.0f;
  Tc[threadIdx.x] = (int)((float)T[threadIdx.x] / ds);
}

// ---------------------------------------------------------------------------
// DFT cos/sin tables, pre-swizzled to WMMA B-operand layout:
// tile (s = K-step over samples, t = bin tile); per lane n = lane&15,
// h = lane>>4; v16h element j -> K = 8h + j (j<8), 16 + 8h + (j-8) (j>=8).
__global__ void k_init_dft(_Float16* __restrict__ cosP, _Float16* __restrict__ sinP) {
  int idx = blockIdx.x * blockDim.x + threadIdx.x;
  if (idx >= KSTEPS * NBT * 512) return;
  int j    = idx & 15;
  int lane = (idx >> 4) & 31;
  int tile = idx >> 9;
  int s = tile / NBT, t = tile % NBT;
  int k   = s * 32 + ((lane < 16) ? 0 : 8) + (j < 8 ? j : j + 8);
  int bin = t * 16 + (lane & 15);
  float cv = 0.f, sv = 0.f;
  if (k < WIN && bin < 200) {
    int ph = (k * bin) % WIN;  // exact phase reduction
    double ang = 6.283185307179586476925286766559 * (double)ph / (double)WIN;
    cv = (float)cos(ang);
    sv = (float)sin(ang);      // sign of Im irrelevant for |X|^2
  }
  cosP[idx] = (_Float16)cv;
  sinP[idx] = (_Float16)sv;
}

// ---------------------------------------------------------------------------
// Mel filterbank table, same B-operand swizzle. bins >= 200 are zero.
__global__ void k_init_banks(_Float16* __restrict__ bnkP) {
  int idx = blockIdx.x * blockDim.x + threadIdx.x;
  if (idx >= SKSTEPS * MELT * 512) return;
  int j    = idx & 15;
  int lane = (idx >> 4) & 31;
  int tile = idx >> 9;
  int s = tile / MELT, u = tile % MELT;
  int bin = s * 32 + ((lane < 16) ? 0 : 8) + (j < 8 ? j : j + 8);
  int m   = u * 16 + (lane & 15);
  float v = 0.f;
  if (bin < 200) {
    double mlow  = 1127.0 * log(1.0 + 20.0   / 700.0);
    double mhigh = 1127.0 * log(1.0 + 8000.0 / 700.0);
    double d    = (mhigh - mlow) / 81.0;
    double left = mlow + (double)m * d;
    double mel  = 1127.0 * log(1.0 + (40.0 * (double)bin) / 700.0);
    double up = (mel - left) / d;
    double dn = (left + 2.0 * d - mel) / d;
    double vv = fmin(up, dn);
    if (vv < 0.0) vv = 0.0;
    v = (float)vv;
  }
  bnkP[idx] = (_Float16)v;
}

// ---------------------------------------------------------------------------
// Main kernel: 1 wave / block, one (batch, 16-frame) tile.
__global__ __launch_bounds__(32) void k_fbank(const float* __restrict__ x,
                                              const float* __restrict__ normalizer,
                                              const _Float16* __restrict__ cosP,
                                              const _Float16* __restrict__ sinP,
                                              const _Float16* __restrict__ bnkP,
                                              float* __restrict__ out) {
  __shared__ __align__(16) _Float16 sF[16 * KP];     // windowed frames (f16)
  __shared__ __align__(16) _Float16 sS[16 * SPECW];  // power spectrum (f16)
  const int  lane = threadIdx.x;
  const int  b    = blockIdx.x / FT_TILES;
  const int  f0   = (blockIdx.x % FT_TILES) * 16;
  const bool hi   = lane >= 16;
  const int  mrow = lane & 15;

  // Phase 0: pre-emphasis + Hann window -> f16 frames in LDS (row major)
  for (int r = 0; r < 16; ++r) {
    int f = f0 + r;
    const float* xf = x + (size_t)b * LEN + (size_t)f * SHIFT;
    for (int i = lane; i < KP; i += 32) {
      _Float16 v = (_Float16)0.f;
      if (f < NFR && i < WIN) {
        float cur = xf[i];
        float prv = (i == 0) ? cur : xf[i - 1];
        float w = 0.5f - 0.5f * __cosf(6.2831853071795864f * (float)i / 399.0f);
        v = (_Float16)((cur - 0.97f * prv) * w);
      }
      sF[r * KP + i] = v;
    }
  }
  // zero spec padding columns 208..223
  for (int p = lane; p < 16 * 16; p += 32)
    sS[(p >> 4) * SPECW + 208 + (p & 15)] = (_Float16)0.f;
  __syncthreads();

  // Phase 1: DFT as WMMA GEMM (re & im), then |X|^2 -> f16 spec in LDS
  for (int t = 0; t < NBT; ++t) {
    v8f cre = {}; v8f cim = {};
    for (int s = 0; s < KSTEPS; ++s) {
      int k0 = s * 32 + (hi ? 8 : 0);
      int k1 = s * 32 + (hi ? 24 : 16);
      union { v16h v; h8 h[2]; } a;
      a.h[0] = *(const h8*)(sF + mrow * KP + k0);
      a.h[1] = *(const h8*)(sF + mrow * KP + k1);
      const v16h bc = *((const v16h*)cosP + (size_t)(s * NBT + t) * 32 + lane);
      const v16h bs = *((const v16h*)sinP + (size_t)(s * NBT + t) * 32 + lane);
      cre = __builtin_amdgcn_wmma_f32_16x16x32_f16(false, a.v, false, bc,
                                                   (short)0, cre, false, false);
      cim = __builtin_amdgcn_wmma_f32_16x16x32_f16(false, a.v, false, bs,
                                                   (short)0, cim, false, false);
    }
    v8f sp = cre * cre + cim * cim;
    int col   = t * 16 + mrow;
    int rbase = hi ? 8 : 0;
#pragma unroll
    for (int r = 0; r < 8; ++r)
      sS[(rbase + r) * SPECW + col] = (_Float16)sp[r];
  }
  __syncthreads();

  // Phase 2: mel GEMM + log + normalize -> global output
  for (int u = 0; u < MELT; ++u) {
    v8f acc = {};
    for (int s = 0; s < SKSTEPS; ++s) {
      int k0 = s * 32 + (hi ? 8 : 0);
      int k1 = s * 32 + (hi ? 24 : 16);
      union { v16h v; h8 h[2]; } a;
      a.h[0] = *(const h8*)(sS + mrow * SPECW + k0);
      a.h[1] = *(const h8*)(sS + mrow * SPECW + k1);
      const v16h bb = *((const v16h*)bnkP + (size_t)(s * MELT + u) * 32 + lane);
      acc = __builtin_amdgcn_wmma_f32_16x16x32_f16(false, a.v, false, bb,
                                                   (short)0, acc, false, false);
    }
    int n = u * 16 + mrow;
    float nrm = normalizer[n];
    int rbase = hi ? 8 : 0;
#pragma unroll
    for (int r = 0; r < 8; ++r) {
      int f = f0 + rbase + r;
      if (f < NFR)
        out[((size_t)b * NFR + f) * NMEL + n] = logf(fmaxf(acc[r], FEPS)) * nrm;
    }
  }
}

// ---------------------------------------------------------------------------
// Masked mean subtraction over the first T_[b] frames, in place.
__global__ void k_meansub(float* __restrict__ out, const int* __restrict__ Tc) {
  int b = blockIdx.x;
  int m = threadIdx.x;  // 0..79, coalesced across the block
  int t = Tc[b];
  float cnt = (float)(t > 0 ? t : 1);
  float sum = 0.f;
  for (int f = 0; f < t; ++f) sum += out[((size_t)b * NFR + f) * NMEL + m];
  float mean = sum / cnt;
  for (int f = 0; f < t; ++f) out[((size_t)b * NFR + f) * NMEL + m] -= mean;
}

// ---------------------------------------------------------------------------
extern "C" void kernel_launch(void* const* d_in, const int* in_sizes, int n_in,
                              void* d_out, int out_size, void* d_ws, size_t ws_size,
                              hipStream_t stream) {
  (void)in_sizes; (void)n_in; (void)out_size; (void)ws_size;
  const float* x          = (const float*)d_in[0];
  const int*   T          = (const int*)d_in[1];
  const float* normalizer = (const float*)d_in[2];
  float* out = (float*)d_out;

  char* ws = (char*)d_ws;
  int*      Tc   = (int*)(ws + WS_T_OFF);
  _Float16* cosP = (_Float16*)(ws + WS_COS_OFF);
  _Float16* sinP = (_Float16*)(ws + WS_SIN_OFF);
  _Float16* bnkP = (_Float16*)(ws + WS_BNK_OFF);

  k_prepT<<<1, 64, 0, stream>>>(T, Tc);
  k_init_dft<<<(KSTEPS * NBT * 512 + 255) / 256, 256, 0, stream>>>(cosP, sinP);
  k_init_banks<<<(SKSTEPS * MELT * 512 + 255) / 256, 256, 0, stream>>>(bnkP);
  k_fbank<<<BATCH * FT_TILES, 32, 0, stream>>>(x, normalizer, cosP, sinP, bnkP, out);
  k_meansub<<<BATCH, NMEL, 0, stream>>>(out, Tc);
}
